// MultiGateMixExperts_38732015075823
// MI455X (gfx1250) — matrix operand order repaired and checked by reading, not compiled, not run on hardware
//
#include <hip/hip_runtime.h>

// ---------------- types ----------------
typedef unsigned short u16t;
typedef __attribute__((ext_vector_type(16))) __bf16        v16bf;
typedef __attribute__((ext_vector_type(8)))  float         v8f;
typedef __attribute__((ext_vector_type(4)))  unsigned int  v4u;

union Frag { v16bf v; v4u q[2]; };   // 32 bytes: one WMMA bf16 A or B operand

__device__ __forceinline__ u16t f2bf(float f) {   // round-to-nearest-even f32 -> bf16
  unsigned u = __float_as_uint(f);
  u += 0x7FFFu + ((u >> 16) & 1u);
  return (u16t)(u >> 16);
}
__device__ __forceinline__ float bf2f(u16t h) {
  return __uint_as_float(((unsigned)h) << 16);
}

constexpr int Bn = 8192, Dn = 1024, Hn = 1024, En = 16;
constexpr int BM = 64;           // batch rows per workgroup
constexpr int NT = 512;          // threads per block (16 waves: 4 M-groups x 4 N-strips)
constexpr int PITCH = 1032;      // bf16 elems per LDS row: 16B-aligned rows, conflict-spread
constexpr int PANEL = 32 * 1024; // bf16 elems in one k-step weight panel (64 KB)

// ---------------- prep: f32 -> bf16 conversions ----------------
__global__ void cvt_x(const float* __restrict__ x, unsigned* __restrict__ xbf, int ndw) {
  int i = blockIdx.x * blockDim.x + threadIdx.x;
  if (i >= ndw) return;
  unsigned lo = f2bf(x[2 * i]);
  unsigned hi = f2bf(x[2 * i + 1]);
  xbf[i] = lo | (hi << 16);
}

// Pack [E,K=1024,N=1024] f32 weights into per-fragment bf16 layout for the WMMA
// B operand (32x16 tile): lane = (k>=16)*16 + n ; dword j = (k&15)>>1 ; low half = even k.
// Fragments for one k-step (kb) are contiguous: 64 frags * 1 KB = one 64 KB panel.
__global__ void pack_w(const float* __restrict__ W, u16t* __restrict__ Wp, long total_dw) {
  long i = (long)blockIdx.x * blockDim.x + threadIdx.x;
  if (i >= total_dw) return;
  unsigned within = (unsigned)(i & 255);           // 256 dwords per 32x16 fragment
  long     frag   = i >> 8;                        // frag = (e*32 + kb)*64 + nb
  unsigned lane = within >> 3, j = within & 7;
  unsigned n  = lane & 15;
  unsigned kk = ((lane >> 4) << 4) + j * 2;
  unsigned nb = (unsigned)(frag & 63);
  unsigned kb = (unsigned)((frag >> 6) & 31);
  unsigned e  = (unsigned)(frag >> 11);
  size_t base = ((size_t)e << 20) + (size_t)(kb * 32 + kk) * 1024 + nb * 16 + n;
  unsigned lo = f2bf(W[base]);
  unsigned hi = f2bf(W[base + 1024]);              // k+1 -> +N stride
  ((unsigned*)Wp)[i] = lo | (hi << 16);
}

// ---------------- one GEMM stage: tile[64,1024] (LDS) x W[1024,1024] -> tile, fused act ----
// B panels staged cooperatively through double-buffered LDS: one 64 KB panel per k-step,
// next panel's global loads in flight across the whole 16-WMMA compute phase.
// ACT: 0 = sigmoid, 1 = relu, 2 = none
template <int ACT>
__device__ __forceinline__ void run_stage(u16t* tile, u16t* pan0, u16t* pan1,
                                          const u16t* __restrict__ Wp,
                                          const float* __restrict__ bias,
                                          int tid, int mg, int colbase, int lane) {
  v8f acc[16] = {};
  const int arow = mg * 16 + (lane & 15);
  const int ksel = (lane >> 4) * 16;               // hi lanes hold K+16..31

  // preload panel for k-step 0 (each thread moves 128 B of the contiguous 64 KB chunk)
  {
    const v4u* src = (const v4u*)Wp + tid * 8;
    v4u*       dst = (v4u*)pan0 + tid * 8;
#pragma unroll
    for (int j = 0; j < 8; ++j) dst[j] = src[j];
  }
  __syncthreads();

  for (int kk = 0; kk < 32; ++kk) {
    u16t* cur = (kk & 1) ? pan1 : pan0;
    u16t* nxt = (kk & 1) ? pan0 : pan1;

    // issue next panel's global loads now; they stay in flight during the WMMAs
    v4u pf[8];
    if (kk + 1 < 32) {
      const v4u* src = (const v4u*)(Wp + (size_t)(kk + 1) * PANEL) + tid * 8;
#pragma unroll
      for (int j = 0; j < 8; ++j) pf[j] = src[j];
    }

    Frag a;                                        // A fragment from LDS tile
    const u16t* ap = tile + arow * PITCH + kk * 32 + ksel;
    a.q[0] = *(const v4u*)(ap);                    // K = +0..7   (hi lanes +16..23)
    a.q[1] = *(const v4u*)(ap + 8);                // K = +8..15  (hi lanes +24..31)
#pragma unroll
    for (int i = 0; i < 16; ++i) {                 // 16 N-tiles: A reused 16x, B from LDS
      Frag b;
      const u16t* bp = cur + (unsigned)((colbase >> 4) + i) * 512 + lane * 16;
      b.q[0] = *(const v4u*)(bp);
      b.q[1] = *(const v4u*)(bp + 8);
      acc[i] = __builtin_amdgcn_wmma_f32_16x16x32_bf16(false, a.v, false, b.v,
                                                       (short)0, acc[i], false, false);
    }

    if (kk + 1 < 32) {                             // commit prefetched panel to LDS
      v4u* dst = (v4u*)nxt + tid * 8;
#pragma unroll
      for (int j = 0; j < 8; ++j) dst[j] = pf[j];
    }
    __syncthreads();                               // panel visible; also fences tile reads
  }

  // last-iteration barrier above => all A reads done; safe to overwrite tile in place
  const int nl = lane & 15;
  const int rbase = mg * 16 + ((lane >> 4) ? 8 : 0);  // C layout: VGPR r -> M=r / M=8+r
#pragma unroll
  for (int i = 0; i < 16; ++i) {
    int col = colbase + i * 16 + nl;
    float bv = bias[col];
#pragma unroll
    for (int r = 0; r < 8; ++r) {
      float v = acc[i][r] + bv;
      if (ACT == 0)      v = 1.f / (1.f + __expf(-v));
      else if (ACT == 1) v = fmaxf(v, 0.f);
      tile[(rbase + r) * PITCH + col] = f2bf(v);
    }
  }
  __syncthreads();
}

// ---------------- fused expert MLP: sigmoid(xW1+b1) -> (relu)@W2+b2 x2 -> dot Wo ----------
__global__ void __launch_bounds__(NT)
experts_kernel(const u16t* __restrict__ xbf, const u16t* __restrict__ W1p,
               const u16t* __restrict__ W2p,
               const float* __restrict__ b1, const float* __restrict__ b2,
               const float* __restrict__ Wo, const float* __restrict__ bo,
               float* __restrict__ f_ws) {
  __shared__ u16t tile[BM * PITCH];                // 129 KB
  __shared__ u16t panel[2 * PANEL];                // 128 KB double-buffered weight panel
  const int e    = blockIdx.y;
  const int m0   = blockIdx.x * BM;
  const int tid  = threadIdx.x;
  const int lane = tid & 31;
  const int w    = tid >> 5;                       // 16 waves: 4 (M) x 4 (N-strips of 256)
  const int mg   = w >> 2;
  const int colbase = (w & 3) * 256;

  // stage the 64x1024 bf16 x-tile into LDS (row-major, padded pitch)
  for (int idx = tid; idx < BM * (Dn / 8); idx += NT) {
    int r = idx >> 7, c = (idx & 127) * 8;
    *(v4u*)&tile[r * PITCH + c] = *(const v4u*)&xbf[(size_t)(m0 + r) * Dn + c];
  }
  __syncthreads();

  const u16t* W1e = W1p + ((size_t)e << 20);
  const u16t* W2e = W2p + ((size_t)e << 20);
  run_stage<0>(tile, panel, panel + PANEL, W1e, b1 + e * Hn, tid, mg, colbase, lane);
  run_stage<1>(tile, panel, panel + PANEL, W2e, b2 + e * Hn, tid, mg, colbase, lane);
  run_stage<2>(tile, panel, panel + PANEL, W2e, b2 + e * Hn, tid, mg, colbase, lane);

  // f[b,e] = h2 . Wo_e + bo_e  (one wave per row, 4 rows per wave)
  const float* Woe = Wo + e * Hn;
#pragma unroll
  for (int rr = 0; rr < 4; ++rr) {
    int r = w + rr * 16;
    float acc = 0.f;
    for (int c = lane; c < Hn; c += 32)
      acc = fmaf(bf2f(tile[r * PITCH + c]), Woe[c], acc);
#pragma unroll
    for (int off = 16; off > 0; off >>= 1) acc += __shfl_xor(acc, off, 32);
    if (lane == 0) f_ws[(size_t)(m0 + r) * En + e] = acc + bo[e];
  }
}

// ---------------- gates: two softmax(D->16) + weighted sums, one wave per row -------------
__global__ void gates_kernel(const float* __restrict__ x,
                             const float* __restrict__ Wg1, const float* __restrict__ bg1,
                             const float* __restrict__ Wg2, const float* __restrict__ bg2,
                             const float* __restrict__ f_ws, float* __restrict__ out) {
  const int lane = threadIdx.x & 31;
  const int w    = threadIdx.x >> 5;
  const int b    = blockIdx.x * 8 + w;
  const int e    = lane & 15;
  const bool hi  = lane >= 16;                     // lanes 0-15: gate1, 16-31: gate2
  const float* Wg = hi ? Wg2 : Wg1;
  const float* xr = x + (size_t)b * Dn;
  float acc = 0.f;
#pragma unroll 4
  for (int d = 0; d < Dn; ++d)
    acc = fmaf(xr[d], Wg[d * En + e], acc);
  acc += hi ? bg2[e] : bg1[e];
  float m = acc;
#pragma unroll
  for (int off = 8; off > 0; off >>= 1) m = fmaxf(m, __shfl_xor(m, off, 16));
  float ex = __expf(acc - m);
  float s = ex;
#pragma unroll
  for (int off = 8; off > 0; off >>= 1) s += __shfl_xor(s, off, 16);
  float contrib = (ex / s) * f_ws[(size_t)b * En + e];
#pragma unroll
  for (int off = 8; off > 0; off >>= 1) contrib += __shfl_xor(contrib, off, 16);
  if (lane == 0)  out[b] = contrib;
  if (lane == 16) out[Bn + b] = contrib;
}

// ---------------- launch ----------------
extern "C" void kernel_launch(void* const* d_in, const int* in_sizes, int n_in,
                              void* d_out, int out_size, void* d_ws, size_t ws_size,
                              hipStream_t stream) {
  (void)in_sizes; (void)n_in; (void)out_size; (void)ws_size;
  const float* x   = (const float*)d_in[0];
  const float* W1  = (const float*)d_in[1];
  const float* b1  = (const float*)d_in[2];
  const float* W2  = (const float*)d_in[3];
  const float* b2  = (const float*)d_in[4];
  const float* Wo  = (const float*)d_in[5];
  const float* bo  = (const float*)d_in[6];
  const float* Wg1 = (const float*)d_in[7];
  const float* bg1 = (const float*)d_in[8];
  const float* Wg2 = (const float*)d_in[9];
  const float* bg2 = (const float*)d_in[10];

  // workspace: xbf 16MB | W1p 32MB | W2p 32MB | f 512KB  (~80.5 MB)
  u16t*  xbf  = (u16t*)d_ws;
  u16t*  W1p  = xbf + (size_t)Bn * Dn;
  u16t*  W2p  = W1p + (size_t)En * Dn * Hn;
  float* f_ws = (float*)(W2p + (size_t)En * Hn * Hn);

  {
    int ndw = Bn * Dn / 2;
    cvt_x<<<(ndw + 255) / 256, 256, 0, stream>>>(x, (unsigned*)xbf, ndw);
  }
  {
    long tdw = (long)En * Dn * Hn / 2;
    int  nb  = (int)((tdw + 255) / 256);
    pack_w<<<nb, 256, 0, stream>>>(W1, W1p, tdw);
    pack_w<<<nb, 256, 0, stream>>>(W2, W2p, tdw);
  }
  dim3 grid(Bn / BM, En);
  experts_kernel<<<grid, NT, 0, stream>>>(xbf, W1p, W2p, b1, b2, Wo, bo, f_ws);
  gates_kernel<<<Bn / 8, 256, 0, stream>>>(x, Wg1, bg1, Wg2, bg2, f_ws, (float*)d_out);
}
